// egcn_42537356099988
// MI455X (gfx1250) — matrix operand order repaired and checked by reading, not compiled, not run on hardware
//
#include <hip/hip_runtime.h>
#include <hip/hip_bf16.h>
#include <math.h>

// ---- problem constants ----
#define B_   8
#define CIN  64
#define CC   128      // 2*Cin (xc channels)
#define CO   64       // Cout
#define CM   32       // Cmid
#define HH   128      // grid row length
#define LL   16384    // H*H
#define LUP  16256    // L - 128 (== len(idx))
#define NP   64       // positions per workgroup tile
// padded LDS pitches (elements) -> conflict-free ds_load_b128 lane->bank maps
#define P128 136      // for 128-wide rows (68 dwords: bank stride 4)
#define P64  72       // for 64-wide rows  (36 dwords)
#define P32  40       // for 32-wide rows  (20 dwords)

typedef __attribute__((ext_vector_type(16))) __bf16 v16bf;
typedef __attribute__((ext_vector_type(8)))  float  v8f;
typedef __attribute__((ext_vector_type(4)))  unsigned int u32x4;
typedef __attribute__((ext_vector_type(4)))  float  v4f;

// ---- bf16 helpers (round-to-nearest-even) ----
__device__ __forceinline__ unsigned short f2bfb(float f) {
  unsigned int u = __builtin_bit_cast(unsigned int, f);
  u = (u + 0x7FFFu + ((u >> 16) & 1u)) >> 16;
  return (unsigned short)u;
}

union U4S { u32x4 q; unsigned short s[8]; };

// ---- WMMA fragment builders (wave32, ISA 7.12.2 layouts), b128 LDS loads ----
// A: 16(M)x32(K) bf16, W stored row-major [row][k] with padded pitch ldw.
// lanes 0-15: M=lane, K {kb=0: 0..7,16..23}; lanes 16-31: M=lane-16, kb=8.
__device__ __forceinline__ v16bf frag_a(const unsigned short* W, int row0, int k0,
                                        int ldw, int lane) {
  int m  = lane & 15;
  int kb = (lane < 16) ? 0 : 8;
  const unsigned short* r = W + (row0 + m) * ldw + k0 + kb;
  union { u32x4 q[2]; v16bf v; } t;
  t.q[0] = *(const u32x4*)(r);        // K = k0+kb   .. +7
  t.q[1] = *(const u32x4*)(r + 16);   // K = k0+kb+16.. +23
  return t.v;
}
// B: 32(K)x16(N) bf16, X stored TRANSPOSED [position][channel] with pitch ldx.
// lanes 0-15: N=lane, K {kb=0: 0..7,16..23}; lanes 16-31: N=lane-16, kb=8.
__device__ __forceinline__ v16bf frag_b(const unsigned short* X, int k0, int col0,
                                        int ldx, int lane) {
  int n  = lane & 15;
  int kb = (lane < 16) ? 0 : 8;
  const unsigned short* c = X + (col0 + n) * ldx + k0 + kb;
  union { u32x4 q[2]; v16bf v; } t;
  t.q[0] = *(const u32x4*)(c);        // K = k0+kb   .. +7
  t.q[1] = *(const u32x4*)(c + 16);   // K = k0+kb+16.. +23
  return t.v;
}
__device__ __forceinline__ v8f wmma_bf16(v16bf a, v16bf b, v8f c) {
  return __builtin_amdgcn_wmma_f32_16x16x32_bf16(false, a, false, b, (short)0, c,
                                                 false, false);
}

// ---- kernel 0: fold BN scale into bf16 weights, build fused biases ----
__global__ __launch_bounds__(256) void prep_kernel(
    const float* Wsq, const float* bsq, const float* gsq, const float* besq,
    const float* msq, const float* vsq,
    const float* Wtr, const float* btr, const float* gtr, const float* betr,
    const float* mtr, const float* vtr,
    const float* Wf1, const float* bf1, const float* gf1, const float* bef1,
    const float* mf1, const float* vf1,
    const float* Wf2, const float* bf2, const float* gf2, const float* bef2,
    const float* mf2, const float* vf2,
    const float* Wf3, const float* bf3,
    unsigned short* wsq, unsigned short* wtr, unsigned short* wf1,
    unsigned short* wf2, unsigned short* wf3, float* biases) {
  int tid = blockIdx.x * blockDim.x + threadIdx.x;
  int nt  = gridDim.x * blockDim.x;
  for (int i = tid; i < CO * CC; i += nt) {
    int o = i >> 7; float s = gsq[o] * rsqrtf(vsq[o] + 1e-5f);
    wsq[i] = f2bfb(Wsq[i] * s);
  }
  for (int i = tid; i < CO * CIN; i += nt) {
    int o = i >> 6; float s = gtr[o] * rsqrtf(vtr[o] + 1e-5f);
    wtr[i] = f2bfb(Wtr[i] * s);
  }
  for (int i = tid; i < CM * CO; i += nt) {
    int o = i >> 6; float s = gf1[o] * rsqrtf(vf1[o] + 1e-5f);
    wf1[i] = f2bfb(Wf1[i] * s);
  }
  for (int i = tid; i < CM * CM; i += nt) {
    int o = i >> 5; float s = gf2[o] * rsqrtf(vf2[o] + 1e-5f);
    wf2[i] = f2bfb(Wf2[i] * s);
  }
  for (int i = tid; i < CO * CM; i += nt) wf3[i] = f2bfb(Wf3[i]);  // f3: no BN
  if (tid < CO) {
    float s = gsq[tid] * rsqrtf(vsq[tid] + 1e-5f);
    biases[tid] = s * (bsq[tid] - msq[tid]) + besq[tid];
    float st = gtr[tid] * rsqrtf(vtr[tid] + 1e-5f);
    biases[64 + tid] = st * (btr[tid] - mtr[tid]) + betr[tid];
    biases[192 + tid] = bf3[tid];
  }
  if (tid < CM) {
    float s1 = gf1[tid] * rsqrtf(vf1[tid] + 1e-5f);
    biases[128 + tid] = s1 * (bf1[tid] - mf1[tid]) + bef1[tid];
    float s2 = gf2[tid] * rsqrtf(vf2[tid] + 1e-5f);
    biases[160 + tid] = s2 * (bf2[tid] - mf2[tid]) + bef2[tid];
  }
}

// ---- kernel 1: softmax over L per (b,c) row -> xc[:,0:64,:] (bf16) ----
__global__ __launch_bounds__(256) void row_softmax(const float* __restrict__ x,
                                                   unsigned short* __restrict__ xc) {
  int row = blockIdx.x;                 // b*64 + c
  int b = row >> 6, c = row & 63;
  const float* src = x + (size_t)row * LL;
  unsigned short* dst = xc + ((size_t)(b * CC + c)) * LL;
  __shared__ float red[256];
  float mx = -INFINITY;
  for (int i = threadIdx.x; i < LL; i += 256) mx = fmaxf(mx, src[i]);
  red[threadIdx.x] = mx; __syncthreads();
  for (int s = 128; s > 0; s >>= 1) {
    if (threadIdx.x < s) red[threadIdx.x] = fmaxf(red[threadIdx.x], red[threadIdx.x + s]);
    __syncthreads();
  }
  mx = red[0]; __syncthreads();
  float sum = 0.f;
  for (int i = threadIdx.x; i < LL; i += 256) sum += __expf(src[i] - mx);
  red[threadIdx.x] = sum; __syncthreads();
  for (int s = 128; s > 0; s >>= 1) {
    if (threadIdx.x < s) red[threadIdx.x] += red[threadIdx.x + s];
    __syncthreads();
  }
  float inv = 1.0f / red[0];
  for (int i = threadIdx.x; i < LL; i += 256) dst[i] = f2bfb(__expf(src[i] - mx) * inv);
}

// ---- kernel 2: softmax over C per (b,l) -> xc[:,64:128,:] (bf16) ----
__global__ __launch_bounds__(256) void col_softmax(const float* __restrict__ x,
                                                   unsigned short* __restrict__ xc) {
  size_t g = (size_t)blockIdx.x * 256 + threadIdx.x;
  if (g >= (size_t)B_ * LL) return;
  int b = (int)(g / LL), l = (int)(g % LL);
  const float* src = x + (size_t)b * CIN * LL + l;
  float mx = -INFINITY;
  for (int c = 0; c < CIN; ++c) mx = fmaxf(mx, src[(size_t)c * LL]);
  float s = 0.f;
  for (int c = 0; c < CIN; ++c) s += __expf(src[(size_t)c * LL] - mx);
  float inv = 1.0f / s;
  unsigned short* dst = xc + ((size_t)b * CC + CIN) * LL + l;
  for (int c = 0; c < CIN; ++c)
    dst[(size_t)c * LL] = f2bfb(__expf(src[(size_t)c * LL] - mx) * inv);
}

// ---- kernel 3: S = relu(bn(W_sq . xc)) via WMMA; emit S + 4 sliced outputs ----
__global__ __launch_bounds__(256) void squeeze_kernel(
    const unsigned short* __restrict__ xc, const unsigned short* __restrict__ wsqW,
    const float* __restrict__ biases, float* __restrict__ S, float* __restrict__ out) {
  __shared__ __align__(16) unsigned short wT[CO * P128];   // 17 KB (padded rows)
  __shared__ __align__(16) unsigned short xT[NP][P128];    // 17 KB (transposed)
  int tid = threadIdx.x, lane = tid & 31, w = tid >> 5;
  int t0 = blockIdx.x * NP;
  int b = t0 / LL, l0 = t0 % LL;
  // weights: 64 rows x 16 chunks of 8 bf16 (b128 copies into padded rows)
  for (int j = tid; j < CO * (CC / 8); j += 256) {
    int r = j >> 4, kc = (j & 15) * 8;
    *(u32x4*)&wT[r * P128 + kc] = *(const u32x4*)&wsqW[r * CC + kc];
  }
  // activations: b128 global loads, transpose-scatter into LDS [p][c]
  const unsigned short* xb = xc + (size_t)b * CC * LL + l0;
  for (int j = tid; j < CC * (NP / 8); j += 256) {
    int cch = j >> 3, p8 = (j & 7) * 8;
    U4S u; u.q = *(const u32x4*)&xb[(size_t)cch * LL + p8];
#pragma unroll
    for (int e = 0; e < 8; ++e) xT[p8 + e][cch] = u.s[e];
  }
  __syncthreads();
  int tA = w, tB = w + 8;
  int oA = (tA & 3) * 16, pA = (tA >> 2) * 16;
  int oB = (tB & 3) * 16, pB = (tB >> 2) * 16;
  v8f accA = {}; v8f accB = {};
#pragma unroll
  for (int k0 = 0; k0 < CC; k0 += 32) {
    accA = wmma_bf16(frag_a(wT, oA, k0, P128, lane), frag_b(&xT[0][0], k0, pA, P128, lane), accA);
    accB = wmma_bf16(frag_a(wT, oB, k0, P128, lane), frag_b(&xT[0][0], k0, pB, P128, lane), accB);
  }
  const size_t O_UP = 8388608, O_MMU = 16711680, O_LO = 25034752, O_MML = 33357824;
  int n = lane & 15, mh = (lane >> 4) * 8;
#pragma unroll
  for (int t = 0; t < 2; ++t) {
#pragma unroll
    for (int r = 0; r < 8; ++r) {
      int o = (t ? oB : oA) + r + mh;
      int l = l0 + (t ? pB : pA) + n;
      float v = fmaxf((t ? accB[r] : accA[r]) + biases[o], 0.f);
      size_t bo = (size_t)(b * CO + o);
      S[bo * LL + l] = v;
      if (l < LUP)               out[O_UP  + bo * LUP + l] = v;           // upper
      if (l >= HH)               out[O_LO  + bo * LUP + (l - HH)] = v;    // lower
      if (((l + 1) & 127) != 0)  out[O_MMU + bo * LUP + (l - (l >> 7))] = v;
      if ((l & 127) != 0)        out[O_MML + bo * LUP + ((l - 1) - ((l - 1) >> 7))] = v;
    }
  }
}

// ---- kernel 4: trans WMMA -> + stencil(S) -> f1 -> f2 -> f3, fused in LDS ----
__global__ __launch_bounds__(256) void fuse_kernel(
    const float* __restrict__ x, const float* __restrict__ S,
    const unsigned short* __restrict__ wtrW, const unsigned short* __restrict__ wf1W,
    const unsigned short* __restrict__ wf2W, const unsigned short* __restrict__ wf3W,
    const float* __restrict__ biases, float* __restrict__ out) {
  __shared__ __align__(16) unsigned short wtrL[CO * P64];  // 9 KB
  __shared__ __align__(16) unsigned short wf1L[CM * P64];  // 4.5 KB
  __shared__ __align__(16) unsigned short wf2L[CM * P32];  // 2.5 KB
  __shared__ __align__(16) unsigned short wf3L[CO * P32];  // 5 KB
  __shared__ __align__(16) float          gnnF[CO][P64];   // 18 KB   [c][p]
  __shared__ __align__(16) unsigned short gnnB[NP][P64];   // 9 KB    [p][c]
  __shared__ __align__(16) union UU {
    unsigned short xf[NP][P64];                                       // 9 KB [p][c]
    struct { unsigned short h1[NP][P32]; unsigned short h2[NP][P32]; } hh; // 10 KB
  } u;
  int tid = threadIdx.x, lane = tid & 31, w = tid >> 5;
  int t0 = blockIdx.x * NP;
  int b = t0 / LL, l0 = t0 % LL;
  // weight copies (b128, into padded rows)
  for (int j = tid; j < CO * (CIN / 8); j += 256) {         // 512
    int r = j >> 3, kc = (j & 7) * 8;
    *(u32x4*)&wtrL[r * P64 + kc] = *(const u32x4*)&wtrW[r * CIN + kc];
  }
  for (int j = tid; j < CM * (CO / 8); j += 256) {          // 256
    int r = j >> 3, kc = (j & 7) * 8;
    *(u32x4*)&wf1L[r * P64 + kc] = *(const u32x4*)&wf1W[r * CO + kc];
  }
  for (int j = tid; j < CM * (CM / 8); j += 256) {          // 128
    int r = j >> 2, kc = (j & 3) * 8;
    *(u32x4*)&wf2L[r * P32 + kc] = *(const u32x4*)&wf2W[r * CM + kc];
  }
  for (int j = tid; j < CO * (CM / 8); j += 256) {          // 256
    int r = j >> 2, kc = (j & 3) * 8;
    *(u32x4*)&wf3L[r * P32 + kc] = *(const u32x4*)&wf3W[r * CM + kc];
  }
  // x tile: float4 global loads, bf16-convert, transpose into LDS [p][c]
  const float* xb = x + (size_t)b * CIN * LL + l0;
  for (int j = tid; j < CIN * (NP / 4); j += 256) {         // 1024
    int cch = j >> 4, p4 = (j & 15) * 4;
    v4f d = *(const v4f*)&xb[(size_t)cch * LL + p4];
#pragma unroll
    for (int e = 0; e < 4; ++e) u.xf[p4 + e][cch] = f2bfb(d[e]);
  }
  __syncthreads();
  { // trans = relu(bn(W_tr . x)), 16 tiles, 2 per wave
    int tA = w, tB = w + 8;
    int oA = (tA & 3) * 16, pA = (tA >> 2) * 16;
    int oB = (tB & 3) * 16, pB = (tB >> 2) * 16;
    v8f aA = {}; v8f aB = {};
#pragma unroll
    for (int k0 = 0; k0 < CIN; k0 += 32) {
      aA = wmma_bf16(frag_a(wtrL, oA, k0, P64, lane), frag_b(&u.xf[0][0], k0, pA, P64, lane), aA);
      aB = wmma_bf16(frag_a(wtrL, oB, k0, P64, lane), frag_b(&u.xf[0][0], k0, pB, P64, lane), aB);
    }
    int n = lane & 15, mh = (lane >> 4) * 8;
#pragma unroll
    for (int r = 0; r < 8; ++r) {
      int oa = oA + r + mh; gnnF[oa][pA + n] = fmaxf(aA[r] + biases[64 + oa], 0.f);
      int ob = oB + r + mh; gnnF[ob][pB + n] = fmaxf(aB[r] + biases[64 + ob], 0.f);
    }
  }
  __syncthreads();
  // gnn = trans + 5-point stencil of S (masked at grid-column boundaries)
  for (int i = tid; i < CO * NP; i += 256) {
    int cch = i >> 6, p = i & 63;       // consecutive threads -> consecutive p (coalesced S)
    int l = l0 + p;
    const float* Sb = S + (size_t)(b * CO + cch) * LL;
    float v = gnnF[cch][p];
    if (l >= HH)              v += Sb[l - HH];   // upper_p
    if ((l & 127) != 0)       v += Sb[l - 1];    // mid_up
    if (l < LL - HH)          v += Sb[l + HH];   // lower_p
    if (((l + 1) & 127) != 0) v += Sb[l + 1];    // mid_lo
    gnnB[p][cch] = f2bfb(v);
  }
  __syncthreads();
  { // h1 = relu(bn(W_f1 . gnn)), 8 tiles, 1 per wave
    int o0 = (w & 1) * 16, p0 = (w >> 1) * 16;
    v8f acc = {};
#pragma unroll
    for (int k0 = 0; k0 < CO; k0 += 32)
      acc = wmma_bf16(frag_a(wf1L, o0, k0, P64, lane), frag_b(&gnnB[0][0], k0, p0, P64, lane), acc);
    int n = lane & 15, mh = (lane >> 4) * 8;
#pragma unroll
    for (int r = 0; r < 8; ++r) {
      int o = o0 + r + mh;
      u.hh.h1[p0 + n][o] = f2bfb(fmaxf(acc[r] + biases[128 + o], 0.f));
    }
  }
  __syncthreads();
  { // h2 = relu(bn(W_f2 . h1)), K=32 single WMMA per tile
    int o0 = (w & 1) * 16, p0 = (w >> 1) * 16;
    v8f acc = {};
    acc = wmma_bf16(frag_a(wf2L, o0, 0, P32, lane), frag_b(&u.hh.h1[0][0], 0, p0, P32, lane), acc);
    int n = lane & 15, mh = (lane >> 4) * 8;
#pragma unroll
    for (int r = 0; r < 8; ++r) {
      int o = o0 + r + mh;
      u.hh.h2[p0 + n][o] = f2bfb(fmaxf(acc[r] + biases[160 + o], 0.f));
    }
  }
  __syncthreads();
  { // out = W_f3 . h2 + b_f3, 16 tiles, 2 per wave
#pragma unroll
    for (int t = 0; t < 2; ++t) {
      int tt = t ? (w + 8) : w;
      int o0 = (tt & 3) * 16, p0 = (tt >> 2) * 16;
      v8f acc = {};
      acc = wmma_bf16(frag_a(wf3L, o0, 0, P32, lane), frag_b(&u.hh.h2[0][0], 0, p0, P32, lane), acc);
      int n = lane & 15, mh = (lane >> 4) * 8;
#pragma unroll
      for (int r = 0; r < 8; ++r) {
        int o = o0 + r + mh;
        int l = l0 + p0 + n;
        out[(size_t)(b * CO + o) * LL + l] = acc[r] + biases[192 + o];
      }
    }
  }
}

extern "C" void kernel_launch(void* const* d_in, const int* in_sizes, int n_in,
                              void* d_out, int out_size, void* d_ws, size_t ws_size,
                              hipStream_t stream) {
  (void)in_sizes; (void)n_in; (void)out_size; (void)ws_size;
  const float* x    = (const float*)d_in[0];
  const float* Wsq  = (const float*)d_in[1];
  const float* bsq  = (const float*)d_in[2];
  const float* gsq  = (const float*)d_in[3];
  const float* besq = (const float*)d_in[4];
  const float* msq  = (const float*)d_in[5];
  const float* vsq  = (const float*)d_in[6];
  const float* Wtr  = (const float*)d_in[7];
  const float* btr  = (const float*)d_in[8];
  const float* gtr  = (const float*)d_in[9];
  const float* betr = (const float*)d_in[10];
  const float* mtr  = (const float*)d_in[11];
  const float* vtr  = (const float*)d_in[12];
  const float* Wf1  = (const float*)d_in[13];
  const float* bf1  = (const float*)d_in[14];
  const float* gf1  = (const float*)d_in[15];
  const float* bef1 = (const float*)d_in[16];
  const float* mf1  = (const float*)d_in[17];
  const float* vf1  = (const float*)d_in[18];
  const float* Wf2  = (const float*)d_in[19];
  const float* bf2  = (const float*)d_in[20];
  const float* gf2  = (const float*)d_in[21];
  const float* bef2 = (const float*)d_in[22];
  const float* mf2  = (const float*)d_in[23];
  const float* vf2  = (const float*)d_in[24];
  const float* Wf3  = (const float*)d_in[25];
  const float* bf3  = (const float*)d_in[26];

  // workspace carve-up (all 16B aligned)
  char* ws = (char*)d_ws;
  unsigned short* xc = (unsigned short*)ws;                   // 33,554,432 B (bf16)
  float* S           = (float*)(ws + 33554432);               // 33,554,432 B (f32)
  unsigned short* wsq = (unsigned short*)(ws + 67108864);     // bf16 weights
  unsigned short* wtr = wsq + CO * CC;
  unsigned short* wf1 = wtr + CO * CIN;
  unsigned short* wf2 = wf1 + CM * CO;
  unsigned short* wf3 = wf2 + CM * CM;
  float* biases = (float*)(wf3 + CO * CM);                    // 256 f32

  float* out = (float*)d_out;

  prep_kernel<<<32, 256, 0, stream>>>(Wsq, bsq, gsq, besq, msq, vsq,
                                      Wtr, btr, gtr, betr, mtr, vtr,
                                      Wf1, bf1, gf1, bef1, mf1, vf1,
                                      Wf2, bf2, gf2, bef2, mf2, vf2,
                                      Wf3, bf3, wsq, wtr, wf1, wf2, wf3, biases);
  row_softmax<<<B_ * CIN, 256, 0, stream>>>(x, xc);
  col_softmax<<<(B_ * LL) / 256, 256, 0, stream>>>(x, xc);
  squeeze_kernel<<<(B_ * LL) / NP, 256, 0, stream>>>(xc, wsq, biases, S, out);
  fuse_kernel<<<(B_ * LL) / NP, 256, 0, stream>>>(x, S, wtr, wf1, wf2, wf3, biases, out);
}
